// GroupedQueryAttention_3315714752696
// MI455X (gfx1250) — matrix operand order repaired and checked by reading, not compiled, not run on hardware
//
#include <hip/hip_runtime.h>
#include <hip/hip_bf16.h>

#define B_   2
#define S_   2048
#define H_   2048
#define NH_  16
#define NKV_ 4
#define D_   128
#define SCALE_ 0.08838834764831845f   // 1/sqrt(128)

typedef __bf16 bf16;
typedef __attribute__((ext_vector_type(16))) __bf16 v16bf;
typedef __attribute__((ext_vector_type(8)))  float  v8f;

union Pack8 { uint4 u; bf16 h[8]; };
union Pack4 { uint2 u; bf16 h[4]; };

static __device__ __forceinline__ int lane_id() { return threadIdx.x & 31; }

static __device__ __forceinline__ v8f zero_v8f() {
  v8f z = {0.f, 0.f, 0.f, 0.f, 0.f, 0.f, 0.f, 0.f};
  return z;
}

static __device__ __forceinline__ v8f wmma_bf16(v16bf a, v16bf b, v8f c) {
  // D = A(16x32 bf16) * B(32x16 bf16) + C(16x16 f32)
  return __builtin_amdgcn_wmma_f32_16x16x32_bf16(
      false, a, false, b, (short)0, c, false, false);
}

// A fragment (16x32, 16-bit): lane<16 -> row=lane, K={0..7,16..23};
// lane>=16 -> row=lane-16, K={8..15,24..31}.
static __device__ __forceinline__ v16bf load_a16x32(const bf16* base, int ldk) {
  const int lane = lane_id();
  const bf16* p = base + (lane & 15) * ldk + ((lane >> 4) << 3);
  v16bf a;
  ((uint4*)&a)[0] = *(const uint4*)(p);
  ((uint4*)&a)[1] = *(const uint4*)(p + 16);
  return a;
}

// B fragment (32x16, 16-bit): lane<16 -> col=lane, K={0..7,16..23}; lane>=16 -> col=lane-16,
// K={8..15,24..31}. Element (k,n) at col_base[n*stride_n + k] (K contiguous per column).
static __device__ __forceinline__ v16bf load_b32x16(const bf16* col_base, int stride_n) {
  const int lane = lane_id();
  const bf16* p = col_base + (lane & 15) * stride_n + ((lane >> 4) << 3);
  v16bf b;
  ((uint4*)&b)[0] = *(const uint4*)(p);
  ((uint4*)&b)[1] = *(const uint4*)(p + 16);
  return b;
}

// ---------------------------------------------------------------------------
// Kernel 0: RoPE cos/sin tables [S][64] -- removes transcendentals from the
// projection epilogue (no __powf Cody-Waite expansion in the hot path).
// ---------------------------------------------------------------------------
__global__ __launch_bounds__(256) void rope_table_kernel(
    float* __restrict__ ct, float* __restrict__ st) {
  int i = blockIdx.x * 256 + threadIdx.x;  // S_*64 items
  int pos = i >> 6, d2 = i & 63;
  // inv_freq = 10000^(-2*d2/128) = exp2(-d2 * log2(10000)/64)
  float invf = __builtin_amdgcn_exp2f(-0.20762050593046015f * (float)d2);
  float a = (float)pos * invf;
  ct[i] = __cosf(a);
  st[i] = __sinf(a);
}

// ---------------------------------------------------------------------------
// Kernel 1: fused QKV projection + RoPE. K-step 64 (8 WMMA between barriers).
// grid.y = 24 head-slots (0..15 Q, 16..19 K, 20..23 V-transposed).
// ---------------------------------------------------------------------------
__global__ __launch_bounds__(256) void qkv_rope_kernel(
    const float* __restrict__ hidden, const float* __restrict__ wq,
    const float* __restrict__ wk, const float* __restrict__ wv,
    const float* __restrict__ costab, const float* __restrict__ sintab,
    bf16* __restrict__ Qb, bf16* __restrict__ Kb, bf16* __restrict__ Vt) {
  __shared__ bf16 As[64 * 64];    // [m][k]
  __shared__ bf16 Bs[128 * 64];   // [n][k]
  __shared__ float Cs[64 * 128];

  const int tid  = threadIdx.x;
  const int wave = tid >> 5;
  const int lane = tid & 31;
  const int row0 = blockIdx.x * 64;
  const int hy   = blockIdx.y;

  const float* W;
  int ldw, ncol0;
  if (hy < NH_)             { W = wq; ldw = NH_ * D_;  ncol0 = hy * D_; }
  else if (hy < NH_ + NKV_) { W = wk; ldw = NKV_ * D_; ncol0 = (hy - NH_) * D_; }
  else                      { W = wv; ldw = NKV_ * D_; ncol0 = (hy - NH_ - NKV_) * D_; }

  const int rw = (wave >> 1) * 16;
  const int cw = (wave & 1) * 64;

  v8f acc[4];
#pragma unroll
  for (int t = 0; t < 4; ++t) acc[t] = zero_v8f();

  const int am = tid >> 2, ak = (tid & 3) * 16;         // A: 16 contiguous k
  const int bn = (tid & 15) * 8, bkq = (tid >> 4) * 4;  // B: 8 cols x 4 k

  for (int k0 = 0; k0 < H_; k0 += 64) {
    { // A tile: fp32 -> packed bf16, ds_store_b128 x2
      const float* src = hidden + (size_t)(row0 + am) * H_ + k0 + ak;
      float4 f[4];
#pragma unroll
      for (int q = 0; q < 4; ++q) f[q] = ((const float4*)src)[q];
      Pack8 p0, p1;
      p0.h[0]=(bf16)f[0].x; p0.h[1]=(bf16)f[0].y; p0.h[2]=(bf16)f[0].z; p0.h[3]=(bf16)f[0].w;
      p0.h[4]=(bf16)f[1].x; p0.h[5]=(bf16)f[1].y; p0.h[6]=(bf16)f[1].z; p0.h[7]=(bf16)f[1].w;
      p1.h[0]=(bf16)f[2].x; p1.h[1]=(bf16)f[2].y; p1.h[2]=(bf16)f[2].z; p1.h[3]=(bf16)f[2].w;
      p1.h[4]=(bf16)f[3].x; p1.h[5]=(bf16)f[3].y; p1.h[6]=(bf16)f[3].z; p1.h[7]=(bf16)f[3].w;
      *(uint4*)(As + am * 64 + ak)     = p0.u;
      *(uint4*)(As + am * 64 + ak + 8) = p1.u;
    }
    { // B tile: 4 k-rows x 8 n, transposed into [n][k]; ds_store_b64 x8
      const float* src = W + (size_t)(k0 + bkq) * ldw + ncol0 + bn;
      float rb[4][8];
#pragma unroll
      for (int rr = 0; rr < 4; ++rr) {
        float4 x = ((const float4*)(src + (size_t)rr * ldw))[0];
        float4 y = ((const float4*)(src + (size_t)rr * ldw))[1];
        rb[rr][0]=x.x; rb[rr][1]=x.y; rb[rr][2]=x.z; rb[rr][3]=x.w;
        rb[rr][4]=y.x; rb[rr][5]=y.y; rb[rr][6]=y.z; rb[rr][7]=y.w;
      }
#pragma unroll
      for (int j = 0; j < 8; ++j) {
        Pack4 pk;
#pragma unroll
        for (int rr = 0; rr < 4; ++rr) pk.h[rr] = (bf16)rb[rr][j];
        *(uint2*)(Bs + (bn + j) * 64 + bkq) = pk.u;
      }
    }
    __syncthreads();
#pragma unroll
    for (int c = 0; c < 2; ++c) {
      v16bf af = load_a16x32(As + rw * 64 + c * 32, 64);
#pragma unroll
      for (int t = 0; t < 4; ++t) {
        v16bf bfr = load_b32x16(Bs + (cw + t * 16) * 64 + c * 32, 64);
        acc[t] = wmma_bf16(af, bfr, acc[t]);
      }
    }
    __syncthreads();
  }

  const int kh = lane >> 4;
#pragma unroll
  for (int t = 0; t < 4; ++t)
#pragma unroll
    for (int r = 0; r < 8; ++r)
      Cs[(rw + r + 8 * kh) * 128 + cw + t * 16 + (lane & 15)] = acc[t][r];
  __syncthreads();

  if (hy < NH_ + NKV_) {
    // Q/K: RoPE from tables, packed global_store_b128 (8 contiguous d)
#pragma unroll
    for (int it = 0; it < 4; ++it) {
      int gi = tid + it * 256;
      int m = gi >> 4, dg = (gi & 15) * 8;
      int gm = row0 + m;
      int b = gm >> 11, pos = gm & (S_ - 1);
      Pack8 pk;
#pragma unroll
      for (int j = 0; j < 8; ++j) {
        int d = dg + j;
        float val     = Cs[m * 128 + d];
        float partner = Cs[m * 128 + (d ^ 64)];
        float cv = costab[pos * 64 + (d & 63)];
        float sv = sintab[pos * 64 + (d & 63)];
        float rot = (d < 64) ? -partner : partner;
        pk.h[j] = (bf16)(val * cv + rot * sv);
      }
      bf16* dst = (hy < NH_)
          ? Qb + (((size_t)b * NH_ + hy) * S_ + pos) * D_ + dg
          : Kb + (((size_t)b * NKV_ + (hy - NH_)) * S_ + pos) * D_ + dg;
      *(uint4*)dst = pk.u;
    }
  } else {
    // V: store transposed [D][S]; 8 contiguous pos per store
    const int vhead = hy - NH_ - NKV_;
    const int b = row0 >> 11, pos0 = row0 & (S_ - 1);
#pragma unroll
    for (int it = 0; it < 4; ++it) {
      int gi = tid + it * 256;
      int d = gi >> 3, mg = (gi & 7) * 8;
      Pack8 pk;
#pragma unroll
      for (int j = 0; j < 8; ++j) pk.h[j] = (bf16)Cs[(mg + j) * 128 + d];
      *(uint4*)(Vt + (((size_t)b * NKV_ + vhead) * D_ + d) * S_ + pos0 + mg) = pk.u;
    }
  }
}

// ---------------------------------------------------------------------------
// Kernel 2: causal flash attention. One wave per 16 query rows; 32-token KV
// tiles; online softmax; P re-layout via per-wave LDS; prefetch of next tile.
// ---------------------------------------------------------------------------
__global__ __launch_bounds__(256) void flash_attn_kernel(
    const bf16* __restrict__ Qb, const bf16* __restrict__ Kb,
    const bf16* __restrict__ Vt, bf16* __restrict__ attnO) {
  __shared__ bf16 ptile_all[8][16 * 32];

  const int tid  = threadIdx.x;
  const int wave = tid >> 5;
  const int lane = tid & 31;
  const int kh   = lane >> 4;
  const int ln15 = lane & 15;

  const int h   = blockIdx.y;
  const int b   = blockIdx.z;
  const int kvh = h >> 2;  // NH/NKV = 4
  const int qrow0 = (blockIdx.x * 8 + wave) * 16;

  const bf16* Qh = Qb + ((size_t)b * NH_ + h) * S_ * D_;
  const bf16* Kh = Kb + ((size_t)b * NKV_ + kvh) * S_ * D_;
  const bf16* Vh = Vt + ((size_t)b * NKV_ + kvh) * (size_t)D_ * S_;
  bf16* pt = ptile_all[wave];

  v16bf qa[4];
#pragma unroll
  for (int cc = 0; cc < 4; ++cc)
    qa[cc] = load_a16x32(Qh + (size_t)qrow0 * D_ + cc * 32, D_);

  v8f o[8];
#pragma unroll
  for (int t = 0; t < 8; ++t) o[t] = zero_v8f();
  float m_i[8], l_i[8];
#pragma unroll
  for (int r = 0; r < 8; ++r) { m_i[r] = -3.0e38f; l_i[r] = 0.0f; }

  for (int j0 = 0; j0 < qrow0 + 16; j0 += 32) {
    // overlap: prefetch next KV tile into cache while this tile computes
    if (j0 + 32 < qrow0 + 16) {
      __builtin_prefetch(Kh + (size_t)(j0 + 32 + ln15) * D_, 0, 0);
      __builtin_prefetch(Vh + (size_t)(ln15 * 8) * S_ + j0 + 32, 0, 0);
    }
    // ---- scores S = Q * K^T ----
    v8f sc[2];
#pragma unroll
    for (int g = 0; g < 2; ++g) {
      v8f s = zero_v8f();
      const bf16* colb = Kh + (size_t)(j0 + g * 16) * D_;
#pragma unroll
      for (int cc = 0; cc < 4; ++cc) {
        v16bf kb = load_b32x16(colb + cc * 32, D_);
        s = wmma_bf16(qa[cc], kb, s);
      }
      const int token = j0 + g * 16 + ln15;
#pragma unroll
      for (int r = 0; r < 8; ++r) {
        float v = s[r] * SCALE_;
        v = (token > qrow0 + 8 * kh + r) ? -3.0e38f : v;  // causal, branch-free
        sc[g][r] = v;
      }
    }
    // ---- online softmax ----
    float mnew[8], alpha[8], psum[8];
#pragma unroll
    for (int r = 0; r < 8; ++r) {
      float rmax = fmaxf(sc[0][r], sc[1][r]);
#pragma unroll
      for (int off = 8; off >= 1; off >>= 1)
        rmax = fmaxf(rmax, __shfl_xor(rmax, off, 32));
      mnew[r]  = fmaxf(m_i[r], rmax);
      alpha[r] = __expf(m_i[r] - mnew[r]);
      m_i[r]   = mnew[r];
      psum[r]  = 0.0f;
    }
#pragma unroll
    for (int g = 0; g < 2; ++g)
#pragma unroll
      for (int r = 0; r < 8; ++r) {
        float p = __expf(sc[g][r] - mnew[r]);
        psum[r] += p;
        pt[(r + 8 * kh) * 32 + g * 16 + ln15] = (bf16)p;
      }
#pragma unroll
    for (int r = 0; r < 8; ++r) {
      float rs = psum[r];
#pragma unroll
      for (int off = 8; off >= 1; off >>= 1)
        rs += __shfl_xor(rs, off, 32);
      l_i[r] = l_i[r] * alpha[r] + rs;
    }
#pragma unroll
    for (int t = 0; t < 8; ++t)
#pragma unroll
      for (int r = 0; r < 8; ++r)
        o[t][r] *= alpha[r];
    asm volatile("s_wait_dscnt 0x0" ::: "memory");
    v16bf pa = load_a16x32(pt, 32);
#pragma unroll
    for (int t = 0; t < 8; ++t) {
      v16bf vb = load_b32x16(Vh + (size_t)(t * 16) * S_ + j0, S_);
      o[t] = wmma_bf16(pa, vb, o[t]);
    }
  }

  float linv[8];
#pragma unroll
  for (int r = 0; r < 8; ++r) linv[r] = 1.0f / l_i[r];
#pragma unroll
  for (int t = 0; t < 8; ++t)
#pragma unroll
    for (int r = 0; r < 8; ++r) {
      int row = qrow0 + r + 8 * kh;
      attnO[((size_t)b * S_ + row) * (NH_ * D_) + h * D_ + t * 16 + ln15] =
          (bf16)(o[t][r] * linv[r]);
    }
}

// ---------------------------------------------------------------------------
// Kernel 3: output projection attnO(bf16) @ Wo -> fp32 out. K-step 64.
// ---------------------------------------------------------------------------
__global__ __launch_bounds__(256) void out_proj_kernel(
    const bf16* __restrict__ attnO, const float* __restrict__ wo,
    float* __restrict__ out) {
  __shared__ bf16 Bs[128 * 64];

  const int tid   = threadIdx.x;
  const int wave  = tid >> 5;
  const int lane  = tid & 31;
  const int row0  = blockIdx.x * 64;
  const int ncol0 = blockIdx.y * 128;
  const int rw = (wave >> 1) * 16;
  const int cw = (wave & 1) * 64;
  const int bn = (tid & 15) * 8, bkq = (tid >> 4) * 4;
  const int ldw = NH_ * D_;  // 2048

  v8f acc[4];
#pragma unroll
  for (int t = 0; t < 4; ++t) acc[t] = zero_v8f();

  for (int k0 = 0; k0 < NH_ * D_; k0 += 64) {
    { // B tile: fp32 -> bf16, transposed [n][k], packed b64 stores
      const float* src = wo + (size_t)(k0 + bkq) * ldw + ncol0 + bn;
      float rb[4][8];
#pragma unroll
      for (int rr = 0; rr < 4; ++rr) {
        float4 x = ((const float4*)(src + (size_t)rr * ldw))[0];
        float4 y = ((const float4*)(src + (size_t)rr * ldw))[1];
        rb[rr][0]=x.x; rb[rr][1]=x.y; rb[rr][2]=x.z; rb[rr][3]=x.w;
        rb[rr][4]=y.x; rb[rr][5]=y.y; rb[rr][6]=y.z; rb[rr][7]=y.w;
      }
#pragma unroll
      for (int j = 0; j < 8; ++j) {
        Pack4 pk;
#pragma unroll
        for (int rr = 0; rr < 4; ++rr) pk.h[rr] = (bf16)rb[rr][j];
        *(uint2*)(Bs + (bn + j) * 64 + bkq) = pk.u;
      }
    }
    __syncthreads();
#pragma unroll
    for (int c = 0; c < 2; ++c) {
      // A fragments straight from global bf16 (row-major matches A striping)
      v16bf af = load_a16x32(attnO + (size_t)(row0 + rw) * ldw + k0 + c * 32, ldw);
#pragma unroll
      for (int t = 0; t < 4; ++t) {
        v16bf bfr = load_b32x16(Bs + (cw + t * 16) * 64 + c * 32, 64);
        acc[t] = wmma_bf16(af, bfr, acc[t]);
      }
    }
    __syncthreads();
  }
  const int kh = lane >> 4;
#pragma unroll
  for (int t = 0; t < 4; ++t)
#pragma unroll
    for (int r = 0; r < 8; ++r)
      out[(size_t)(row0 + rw + r + 8 * kh) * ldw + ncol0 + cw + t * 16 + (lane & 15)] =
          acc[t][r];
}

// ---------------------------------------------------------------------------
extern "C" void kernel_launch(void* const* d_in, const int* in_sizes, int n_in,
                              void* d_out, int out_size, void* d_ws, size_t ws_size,
                              hipStream_t stream) {
  (void)in_sizes; (void)n_in; (void)out_size; (void)ws_size;
  const float* hidden = (const float*)d_in[0];
  const float* wq = (const float*)d_in[1];
  const float* wk = (const float*)d_in[2];
  const float* wv = (const float*)d_in[3];
  const float* wo = (const float*)d_in[4];
  float* out = (float*)d_out;

  char* ws = (char*)d_ws;
  const size_t qb_bytes  = (size_t)B_ * NH_ * S_ * D_ * sizeof(bf16);   // 16 MB
  const size_t kb_bytes  = (size_t)B_ * NKV_ * S_ * D_ * sizeof(bf16);  // 4 MB
  const size_t ao_bytes  = qb_bytes;                                    // 16 MB
  const size_t tab_bytes = (size_t)S_ * 64 * sizeof(float);             // 512 KB
  bf16*  Qb     = (bf16*)ws;
  bf16*  Kb     = (bf16*)(ws + qb_bytes);
  bf16*  Vt     = (bf16*)(ws + qb_bytes + kb_bytes);
  bf16*  attnO  = (bf16*)(ws + qb_bytes + 2 * kb_bytes);
  float* costab = (float*)(ws + qb_bytes + 2 * kb_bytes + ao_bytes);
  float* sintab = (float*)(ws + qb_bytes + 2 * kb_bytes + ao_bytes + tab_bytes);

  rope_table_kernel<<<(S_ * 64) / 256, 256, 0, stream>>>(costab, sintab);

  dim3 g1((B_ * S_) / 64, NH_ + 2 * NKV_, 1);
  qkv_rope_kernel<<<g1, 256, 0, stream>>>(hidden, wq, wk, wv, costab, sintab,
                                          Qb, Kb, Vt);

  dim3 g2(S_ / 128, NH_, B_);
  flash_attn_kernel<<<g2, 256, 0, stream>>>(Qb, Kb, Vt, attnO);

  dim3 g3((B_ * S_) / 64, H_ / 128, 1);
  out_proj_kernel<<<g3, 256, 0, stream>>>(attnO, wo, out);
}
